// Model_40630390620620
// MI455X (gfx1250) — matrix operand order repaired and checked by reading, compiled, not run on hardware
//
#include <hip/hip_runtime.h>

// ---------- problem dims ----------
#define BB 8
#define TT 1024
#define CC 768
#define HH 12
#define HD 64
#define N3C 2304                       // 3*C
#define HEADSZ ((size_t)BB*HH*TT*HD)   // elements per q/k/v tensor in ws
#define XSZ   ((size_t)BB*TT*CC)       // x elements
#define WSZ   ((size_t)N3C*CC)         // weight elements

typedef __attribute__((ext_vector_type(16))) __bf16 v16bf;
typedef __attribute__((ext_vector_type(8)))  float  v8f;
typedef int b128t __attribute__((vector_size(16)));   // int4 vector for async builtin

#define AS1 __attribute__((address_space(1)))
#define AS3 __attribute__((address_space(3)))

#if defined(__has_builtin)
#  if __has_builtin(__builtin_amdgcn_global_load_async_to_lds_b128)
#    define HAVE_ASYNC 1
#  endif
#endif
#ifndef HAVE_ASYNC
#  define HAVE_ASYNC 0
#endif

union Frag16 { v16bf v; uint4 q[2]; };
union Pack8  { __bf16 h[8];  uint4 q; };

__device__ __forceinline__ v8f zero8() { v8f z; for (int i = 0; i < 8; ++i) z[i] = 0.f; return z; }

__device__ __forceinline__ void cvt4(__bf16* d, float4 f) {
    d[0] = (__bf16)f.x; d[1] = (__bf16)f.y; d[2] = (__bf16)f.z; d[3] = (__bf16)f.w;
}

// async (or fallback sync) copy of 8 bf16 (128b) global -> LDS
__device__ __forceinline__ void async_copy16(const __bf16* g, __bf16* l) {
#if HAVE_ASYNC
    __builtin_amdgcn_global_load_async_to_lds_b128((AS1 b128t*)(g), (AS3 b128t*)(l), 0, 0);
#else
    *(uint4*)l = *(const uint4*)g;
#endif
}

template <int N>
__device__ __forceinline__ void wait_async_n() {
#if HAVE_ASYNC
#  if __has_builtin(__builtin_amdgcn_s_wait_asynccnt)
    __builtin_amdgcn_s_wait_asynccnt(N);
#  else
    asm volatile("s_wait_asynccnt %0" :: "i"(N));
#  endif
#endif
}

__device__ __forceinline__ v16bf load_frag(const __bf16* p0, const __bf16* p1) {
    Frag16 f;
    f.q[0] = *(const uint4*)p0;
    f.q[1] = *(const uint4*)p1;
    return f.v;
}

__device__ __forceinline__ v8f wmma_bf16(v16bf a, v16bf b, v8f c) {
    return __builtin_amdgcn_wmma_f32_16x16x32_bf16(
        false, a, false, b, (short)0, c, false, false);
}

// =====================================================================
// Kernel 0: fp32 -> bf16 convert (one shot; removes per-tile re-convert)
// =====================================================================
__global__ __launch_bounds__(256)
void cvt_kernel(const float* __restrict__ src, __bf16* __restrict__ dst, int n)
{
    const int i = (blockIdx.x * 256 + threadIdx.x) * 8;
    if (i + 8 <= n) {
        Pack8 t;
        cvt4(t.h + 0, *(const float4*)(src + i));
        cvt4(t.h + 4, *(const float4*)(src + i + 4));
        *(uint4*)(dst + i) = t.q;
    }
}

// =====================================================================
// Kernel 1: qkv = Xbf(8192x768) @ Wbf^T + bias  -> bf16 ws
//   q,k stored (b,h,t,d); v stored TRANSPOSED (b,h,d,t) for attention.
// 256 thr / 8 waves, tile 128x128, K-step 32, double-buffered async LDS.
// =====================================================================
__global__ __launch_bounds__(256)
void qkv_gemm_kernel(const __bf16* __restrict__ Xbf, const __bf16* __restrict__ Wbf,
                     const float* __restrict__ bias, __bf16* __restrict__ qkv)
{
    __shared__ __align__(16) __bf16 Ash[2][128 * 32];   // 16 KB
    __shared__ __align__(16) __bf16 Bsh[2][128 * 32];   // 16 KB

    const int tid  = threadIdx.x;
    const int lane = tid & 31;
    const int wv   = tid >> 5;
    const int mb   = blockIdx.y * 128;
    const int nb   = blockIdx.x * 128;

    v8f acc[8];
    for (int n = 0; n < 8; ++n) acc[n] = zero8();

    const int sr = tid >> 1;             // staged row 0..127 (one b128 chunk each)
    const int sp = (tid & 1) * 16;       // k-part 0/16

    auto stage = [&](int buf, int k0) {
        async_copy16(Xbf + (size_t)(mb + sr) * CC + k0 + sp, &Ash[buf][sr * 32 + sp]);
        async_copy16(Wbf + (size_t)(nb + sr) * CC + k0 + sp, &Bsh[buf][sr * 32 + sp]);
    };

    stage(0, 0);
    const int nk = CC / 32;              // 24
    for (int kt = 0; kt < nk; ++kt) {
        const int cur = kt & 1;
        if (kt + 1 < nk) { stage(cur ^ 1, (kt + 1) * 32); wait_async_n<2>(); }
        else             { wait_async_n<0>(); }
        __syncthreads();

        const int arow = wv * 16 + (lane & 15);
        const int kb   = (lane >> 4) * 8;
        v16bf af = load_frag(&Ash[cur][arow * 32 + kb], &Ash[cur][arow * 32 + 16 + kb]);

        for (int n = 0; n < 8; ++n) {
            const int brow = n * 16 + (lane & 15);
            const int kbb  = (lane >> 4) * 8;
            v16bf bf = load_frag(&Bsh[cur][brow * 32 + kbb], &Bsh[cur][brow * 32 + kbb + 16]);
            acc[n] = wmma_bf16(af, bf, acc[n]);
        }
        __syncthreads();
    }

    // ---- epilogue: +bias, scatter; which/h/dbase are UNIFORM per n-chunk ----
    const int mrow = mb + wv * 16 + (lane < 16 ? 0 : 8);
    const int ln   = lane & 15;
    for (int n = 0; n < 8; ++n) {
        const int   base  = nb + n * 16;        // uniform, multiple of 16
        const int   which = base / CC;          // uniform (16-window never straddles)
        const int   cb    = base % CC;
        const int   hh    = cb >> 6;            // uniform
        const int   d     = (cb & 63) + ln;
        const float bv    = bias[base + ln];
        if (which == 2) {                       // V: transposed (b,h,d,t)
            for (int i = 0; i < 8; ++i) {
                const int row = mrow + i, bi = row >> 10, t = row & 1023;
                qkv[2 * HEADSZ + ((size_t)(bi * HH + hh) * HD + d) * TT + t] =
                    (__bf16)(acc[n][i] + bv);
            }
        } else {                                // Q/K: (b,h,t,d)
            for (int i = 0; i < 8; ++i) {
                const int row = mrow + i, bi = row >> 10, t = row & 1023;
                qkv[(size_t)which * HEADSZ + ((size_t)(bi * HH + hh) * TT + t) * HD + d] =
                    (__bf16)(acc[n][i] + bv);
            }
        }
    }
}

// =====================================================================
// Kernel 2: per (b,h,128 q-rows): Y = relu_mask(Q K^T / 8) @ V
// KV tiles of 32, double-buffered async LDS staging, causal bound uniform.
// =====================================================================
__global__ __launch_bounds__(256)
void attn_kernel(const __bf16* __restrict__ qkv, float* __restrict__ out)
{
    __shared__ __align__(16) __bf16 Ksh[2][32 * 64];   // [kv][d]      8 KB
    __shared__ __align__(16) __bf16 Vsh[2][64 * 40];   // [d][kv] pad  10 KB
    __shared__ __align__(16) __bf16 Ssh[8][16 * 32];   // per-wave S   8 KB

    const int tid  = threadIdx.x;
    const int lane = tid & 31;
    const int wv   = tid >> 5;
    const int q0   = blockIdx.x * 128;
    const int h    = blockIdx.y;
    const int b    = blockIdx.z;

    const size_t hb = (size_t)(b * HH + h);
    const __bf16* gq  = qkv + hb * TT * HD;
    const __bf16* gk  = qkv + HEADSZ + hb * TT * HD;
    const __bf16* gvT = qkv + 2 * HEADSZ + hb * HD * TT;   // [d][t]

    // ---- Q fragments (16x64 as two 16x32 A-chunks) ----
    v16bf qf[2];
    {
        const int row = q0 + wv * 16 + (lane & 15);
        const int kb  = (lane >> 4) * 8;
        for (int c2 = 0; c2 < 2; ++c2) {
            const __bf16* p = gq + (size_t)row * HD + c2 * 32;
            qf[c2] = load_frag(p + kb, p + 16 + kb);
        }
    }

    v8f accy[4];
    for (int dc = 0; dc < 4; ++dc) accy[dc] = zero8();
    const float scale = 0.125f;

    // stager: waves 0-3 -> K tile (128 chunks), waves 4-7 -> V tile (128 chunks)
    auto stage = [&](int buf, int j0) {
        if (tid < 128) {
            const int r = tid >> 2, part = (tid & 3) * 16;
            async_copy16(gk + (size_t)(j0 + r) * HD + part, &Ksh[buf][r * 64 + part]);
        } else {
            const int c2 = tid - 128;
            const int d = c2 >> 1, half = (c2 & 1) * 16;
            async_copy16(gvT + (size_t)d * TT + j0 + half, &Vsh[buf][d * 40 + half]);
        }
    };

    const int ntiles = (q0 + 128) >> 5;     // causal frontier, uniform
    stage(0, 0);
    for (int jt = 0; jt < ntiles; ++jt) {
        const int cur = jt & 1;
        const int j0  = jt << 5;
        if (jt + 1 < ntiles) { stage(cur ^ 1, (jt + 1) << 5); wait_async_n<1>(); }
        else                 { wait_async_n<0>(); }
        __syncthreads();

        // ---- S = Q @ K^T ----
        v8f accs[2];
        accs[0] = zero8(); accs[1] = zero8();
        for (int nc = 0; nc < 2; ++nc) {
            const int kvrow = nc * 16 + (lane & 15);
            for (int kd = 0; kd < 2; ++kd) {
                const int dbase = kd * 32 + (lane >> 4) * 8;
                v16bf bfv = load_frag(&Ksh[cur][kvrow * 64 + dbase],
                                      &Ksh[cur][kvrow * 64 + dbase + 16]);
                accs[nc] = wmma_bf16(qf[kd], bfv, accs[nc]);
            }
        }

        // ---- scale, causal mask, relu; spill bf16 to per-wave LDS ----
        const int rloc = (lane < 16 ? 0 : 8);
        const int rowg = q0 + wv * 16 + rloc;
        for (int nc = 0; nc < 2; ++nc) {
            const int colg = j0 + nc * 16 + (lane & 15);
            for (int i = 0; i < 8; ++i) {
                float v = accs[nc][i] * scale;
                v = (colg <= rowg + i) ? (v > 0.f ? v : 0.f) : 0.f;
                Ssh[wv][(rloc + i) * 32 + nc * 16 + (lane & 15)] = (__bf16)v;
            }
        }

        // ---- reload S as A-fragment (same-wave DS ordering) ----
        const int srow = lane & 15;
        const int skb  = (lane >> 4) * 8;
        v16bf sf = load_frag(&Ssh[wv][srow * 32 + skb],
                             &Ssh[wv][srow * 32 + 16 + skb]);

        // ---- Y += S @ V ----
        for (int dc = 0; dc < 4; ++dc) {
            const int nd  = dc * 16 + (lane & 15);
            const int kvb = (lane >> 4) * 8;
            v16bf vf = load_frag(&Vsh[cur][nd * 40 + kvb],
                                 &Vsh[cur][nd * 40 + kvb + 16]);
            accy[dc] = wmma_bf16(sf, vf, accy[dc]);
        }
        __syncthreads();
    }

    // ---- write Y fp32 to (B,T,C) ----
    for (int dc = 0; dc < 4; ++dc) {
        const int d = dc * 16 + (lane & 15);
        for (int i = 0; i < 8; ++i) {
            const int t = q0 + wv * 16 + (lane < 16 ? 0 : 8) + i;
            out[((size_t)b * TT + t) * CC + h * HD + d] = accy[dc][i];
        }
    }
}

// =====================================================================
extern "C" void kernel_launch(void* const* d_in, const int* in_sizes, int n_in,
                              void* d_out, int out_size, void* d_ws, size_t ws_size,
                              hipStream_t stream)
{
    (void)in_sizes; (void)n_in; (void)out_size; (void)ws_size;
    const float* x    = (const float*)d_in[0];   // (8,1024,768)
    const float* w    = (const float*)d_in[1];   // (2304,768)
    const float* bias = (const float*)d_in[2];   // (2304,)
    float* out = (float*)d_out;

    __bf16* qkv = (__bf16*)d_ws;                 // 3*HEADSZ bf16
    __bf16* Xbf = qkv + 3 * HEADSZ;
    __bf16* Wbf = Xbf + XSZ;

    cvt_kernel<<<dim3((int)(XSZ / 2048)), 256, 0, stream>>>(x, Xbf, (int)XSZ);
    cvt_kernel<<<dim3((int)(WSZ / 2048)), 256, 0, stream>>>(w, Wbf, (int)WSZ);
    qkv_gemm_kernel<<<dim3(N3C / 128, (BB * TT) / 128), 256, 0, stream>>>(Xbf, Wbf, bias, qkv);
    attn_kernel<<<dim3(TT / 128, HH, BB), 256, 0, stream>>>(qkv, out);
}
